// Pointnet2SegHead_29454885716544
// MI455X (gfx1250) — matrix operand order repaired and checked by reading, not compiled, not run on hardware
//
#include <hip/hip_runtime.h>

typedef __attribute__((ext_vector_type(16))) __bf16 v16bf;
typedef __attribute__((ext_vector_type(8)))  float  v8f;
typedef __attribute__((ext_vector_type(4)))  unsigned int u32x4;
typedef unsigned short u16;

union Frag16 { v16bf bf; u32x4 q[2]; };

__device__ __forceinline__ u16 f2bf(float f) {
    unsigned int u = __float_as_uint(f);
    unsigned int r = u + 0x7FFFu + ((u >> 16) & 1u);
    return (u16)(r >> 16);
}
__device__ __forceinline__ float bf2f(u16 h) {
    return __uint_as_float(((unsigned int)h) << 16);
}

// ---------------------------------------------------------------------------
// zero a float buffer
__global__ void zero_kernel(float* p, int n) {
    int i = blockIdx.x * blockDim.x + threadIdx.x;
    if (i < n) p[i] = 0.f;
}

// ---------------------------------------------------------------------------
// fp32 weights (O,Cin) -> bf16 (O,Cpad), zero pad Cin..Cpad
__global__ void wcvt_kernel(const float* __restrict__ Ws, u16* __restrict__ Wd,
                            int O, int Cin, int Cpad) {
    int id = blockIdx.x * blockDim.x + threadIdx.x;
    if (id >= O * Cpad) return;
    int c = id % Cpad, o = id / Cpad;
    float v = (c < Cin) ? Ws[o * Cin + c] : 0.f;
    Wd[id] = f2bf(v);
}

// ---------------------------------------------------------------------------
// three-NN interpolate #1: unknown=sa1 (2048/b), known=sa2 (512/b),
// feats=backbone (B,256,512) fp32 channel-major; extra=sa1_features (B,128,2048).
// Output X1 bf16 [B*2048, 384].
__global__ void interp1_kernel(const float* __restrict__ uxyz,
                               const float* __restrict__ kxyz,
                               const float* __restrict__ kfeat,
                               const float* __restrict__ extra,
                               u16* __restrict__ Xout) {
    const int n = 2048, m = 512, C = 256, Ce = 128, Cpad = 384;
    int b = blockIdx.y;
    __shared__ float sk[512 * 3];
    for (int i = threadIdx.x; i < m * 3; i += blockDim.x)
        sk[i] = kxyz[(size_t)b * m * 3 + i];
    __syncthreads();
    int pn = blockIdx.x * blockDim.x + threadIdx.x;
    const float* up = uxyz + ((size_t)b * n + pn) * 3;
    float ux = up[0], uy = up[1], uz = up[2];
    float d0 = 3.4e38f, d1 = 3.4e38f, d2 = 3.4e38f;
    int i0 = 0, i1 = 0, i2 = 0;
    for (int j = 0; j < m; ++j) {
        float dx = ux - sk[j * 3 + 0];
        float dy = uy - sk[j * 3 + 1];
        float dz = uz - sk[j * 3 + 2];
        float d = dx * dx + dy * dy + dz * dz;
        if (d < d0)      { d2 = d1; i2 = i1; d1 = d0; i1 = i0; d0 = d; i0 = j; }
        else if (d < d1) { d2 = d1; i2 = i1; d1 = d;  i1 = j; }
        else if (d < d2) { d2 = d;  i2 = j; }
    }
    float r0 = 1.f / (sqrtf(fmaxf(d0, 0.f)) + 1e-8f);
    float r1 = 1.f / (sqrtf(fmaxf(d1, 0.f)) + 1e-8f);
    float r2 = 1.f / (sqrtf(fmaxf(d2, 0.f)) + 1e-8f);
    float rs = 1.f / (r0 + r1 + r2);
    float w0 = r0 * rs, w1 = r1 * rs, w2 = r2 * rs;
    const float* F = kfeat + (size_t)b * C * m;
    size_t xrow = ((size_t)b * n + pn) * Cpad;
    for (int c = 0; c < C; ++c) {
        float v = w0 * F[c * m + i0] + w1 * F[c * m + i1] + w2 * F[c * m + i2];
        Xout[xrow + c] = f2bf(v);
    }
    const float* E = extra + (size_t)b * Ce * n;
    for (int c = 0; c < Ce; ++c)
        Xout[xrow + C + c] = f2bf(E[(size_t)c * n + pn]);
}

// ---------------------------------------------------------------------------
// three-NN interpolate #2: unknown=input (8192/b), known=sa1 (2048/b),
// feats=f1 bf16 [B*2048,256] point-major; extra=input_features (B,3,8192).
// Output X2 bf16 [B*8192, 288] (channels 259..287 zero).
__global__ void interp2_kernel(const float* __restrict__ uxyz,
                               const float* __restrict__ kxyz,
                               const u16* __restrict__ f1,
                               const float* __restrict__ extra,
                               u16* __restrict__ Xout) {
    const int n = 8192, m = 2048, C = 256, Cpad = 288;
    int b = blockIdx.y;
    __shared__ float sk[2048 * 3];
    for (int i = threadIdx.x; i < m * 3; i += blockDim.x)
        sk[i] = kxyz[(size_t)b * m * 3 + i];
    __syncthreads();
    int pn = blockIdx.x * blockDim.x + threadIdx.x;
    const float* up = uxyz + ((size_t)b * n + pn) * 3;
    float ux = up[0], uy = up[1], uz = up[2];
    float d0 = 3.4e38f, d1 = 3.4e38f, d2 = 3.4e38f;
    int i0 = 0, i1 = 0, i2 = 0;
    for (int j = 0; j < m; ++j) {
        float dx = ux - sk[j * 3 + 0];
        float dy = uy - sk[j * 3 + 1];
        float dz = uz - sk[j * 3 + 2];
        float d = dx * dx + dy * dy + dz * dz;
        if (d < d0)      { d2 = d1; i2 = i1; d1 = d0; i1 = i0; d0 = d; i0 = j; }
        else if (d < d1) { d2 = d1; i2 = i1; d1 = d;  i1 = j; }
        else if (d < d2) { d2 = d;  i2 = j; }
    }
    float r0 = 1.f / (sqrtf(fmaxf(d0, 0.f)) + 1e-8f);
    float r1 = 1.f / (sqrtf(fmaxf(d1, 0.f)) + 1e-8f);
    float r2 = 1.f / (sqrtf(fmaxf(d2, 0.f)) + 1e-8f);
    float rs = 1.f / (r0 + r1 + r2);
    float w0 = r0 * rs, w1 = r1 * rs, w2 = r2 * rs;
    const u16* F0 = f1 + ((size_t)b * m + i0) * C;
    const u16* F1 = f1 + ((size_t)b * m + i1) * C;
    const u16* F2 = f1 + ((size_t)b * m + i2) * C;
    size_t xrow = ((size_t)b * n + pn) * Cpad;
    // vectorized gather: 8 bf16 channels per b128 load per source row
    for (int c = 0; c < C; c += 8) {
        u32x4 q0 = *reinterpret_cast<const u32x4*>(F0 + c);
        u32x4 q1 = *reinterpret_cast<const u32x4*>(F1 + c);
        u32x4 q2 = *reinterpret_cast<const u32x4*>(F2 + c);
        u32x4 res;
#pragma unroll
        for (int j = 0; j < 4; ++j) {
            float lo = w0 * bf2f((u16)(q0[j] & 0xffffu)) +
                       w1 * bf2f((u16)(q1[j] & 0xffffu)) +
                       w2 * bf2f((u16)(q2[j] & 0xffffu));
            float hi = w0 * bf2f((u16)(q0[j] >> 16)) +
                       w1 * bf2f((u16)(q1[j] >> 16)) +
                       w2 * bf2f((u16)(q2[j] >> 16));
            res[j] = ((unsigned int)f2bf(hi) << 16) | (unsigned int)f2bf(lo);
        }
        *reinterpret_cast<u32x4*>(&Xout[xrow + c]) = res;
    }
    const float* E = extra + (size_t)b * 3 * n;
    for (int c = 0; c < 3; ++c)
        Xout[xrow + C + c] = f2bf(E[(size_t)c * n + pn]);
    for (int c = C + 3; c < Cpad; ++c) Xout[xrow + c] = 0;
}

// ---------------------------------------------------------------------------
// bf16 WMMA GEMM, 4-way output-channel register blocking, compile-time K:
// Y[p,o] = sum_c Wb[o,c] * X[p,c] + bias[o]
// X: [P, CPAD] bf16 row-major; Wb: [256, CPAD] bf16; Y: [P, 256] fp32.
// Block = 256 threads (8 waves). Wave w: ptile = blockIdx.x*2 + (w>>2),
// output channels [ (w&3)*64, (w&3)*64+64 ) as 4 16x16 tiles sharing one
// B fragment per K-step. CPAD templated so the K loop fully unrolls and the
// scheduler can hoist loads across K-steps (staggered s_wait_loadcnt).
template <int CPAD>
__global__ void gemm_wmma_kernel(const u16* __restrict__ X,
                                 const u16* __restrict__ Wb,
                                 const float* __restrict__ bias,
                                 float* __restrict__ Y) {
    const int O = 256;
    int lane = threadIdx.x & 31;
    int wave = threadIdx.x >> 5;
    int ogrp  = (wave & 3) * 64;
    int pbase = (blockIdx.x * 2 + (wave >> 2)) * 16;
    int h   = lane >> 4;   // half-wave: 0 or 1
    int l16 = lane & 15;
    // B (activations): lane col n = l16; K chunks h*8 and 16+h*8
    const u16* brow = X + (size_t)(pbase + l16) * CPAD + h * 8;
    // A (weights): lane row m = l16 within each otile; K chunks h*8 and 16+h*8
    const u16* arow = Wb + (size_t)(ogrp + l16) * CPAD + h * 8;
    v8f acc[4] = {{}, {}, {}, {}};
#pragma unroll
    for (int kb = 0; kb < CPAD; kb += 32) {
        Frag16 b;
        b.q[0] = *reinterpret_cast<const u32x4*>(brow + kb);
        b.q[1] = *reinterpret_cast<const u32x4*>(brow + kb + 16);
#pragma unroll
        for (int t = 0; t < 4; ++t) {
            Frag16 a;
            a.q[0] = *reinterpret_cast<const u32x4*>(arow + t * (16 * CPAD) + kb);
            a.q[1] = *reinterpret_cast<const u32x4*>(arow + t * (16 * CPAD) + kb + 16);
            acc[t] = __builtin_amdgcn_wmma_f32_16x16x32_bf16(
                false, a.bf, false, b.bf, (short)0, acc[t], false, false);
        }
    }
    // D layout: lane l16 = column p, rows o = obase + 8*h + r (r=0..7) contiguous
#pragma unroll
    for (int t = 0; t < 4; ++t) {
        int obase = ogrp + t * 16;
        const float* bp = bias + obase + h * 8;
        float* yp = Y + (size_t)(pbase + l16) * O + obase + h * 8;
#pragma unroll
        for (int r = 0; r < 8; ++r) yp[r] = acc[t][r] + bp[r];
    }
}

// ---------------------------------------------------------------------------
// per-channel sum / sumsq of Y [P,256] into st[0..255]=sum, st[256..511]=sumsq
__global__ void stats_kernel(const float* __restrict__ Y, float* __restrict__ st, int P) {
    int o = threadIdx.x;  // 256 threads
    float s = 0.f, ss = 0.f;
    for (int p = blockIdx.x; p < P; p += gridDim.x) {
        float v = Y[(size_t)p * 256 + o];
        s += v; ss += v * v;
    }
    atomicAdd(&st[o], s);
    atomicAdd(&st[256 + o], ss);
}

// ---------------------------------------------------------------------------
// BN (batch stats) + ReLU, write bf16 [P,256]
__global__ void bnrelu_bf16_kernel(const float* __restrict__ Y, const float* __restrict__ st,
                                   const float* __restrict__ g, const float* __restrict__ be,
                                   u16* __restrict__ Xn, float invP) {
    size_t id = (size_t)blockIdx.x * blockDim.x + threadIdx.x;
    int o = (int)(id & 255);
    float m = st[o] * invP;
    float v = st[256 + o] * invP - m * m;
    float r = (Y[id] - m) * rsqrtf(v + 1e-5f) * g[o] + be[o];
    Xn[id] = f2bf(fmaxf(r, 0.f));
}

// ---------------------------------------------------------------------------
// compute f2 = relu(bn(Y)) on the fly and accumulate its sum/sumsq (no store)
__global__ void bnrelu_stats_kernel(const float* __restrict__ Y, const float* __restrict__ stY,
                                    const float* __restrict__ g, const float* __restrict__ be,
                                    float* __restrict__ stOut, int P, float invP) {
    int o = threadIdx.x;
    float m = stY[o] * invP;
    float v = stY[256 + o] * invP - m * m;
    float sc = rsqrtf(v + 1e-5f) * g[o];
    float bb = be[o] - m * sc;
    float s = 0.f, ss = 0.f;
    for (int p = blockIdx.x; p < P; p += gridDim.x) {
        float f = fmaxf(Y[(size_t)p * 256 + o] * sc + bb, 0.f);
        s += f; ss += f * f;
    }
    atomicAdd(&stOut[o], s);
    atomicAdd(&stOut[256 + o], ss);
}

// ---------------------------------------------------------------------------
// final: f2 = relu(bn2b(Y2b)); out = relu(bnF(f2)*gn+ben), layout (B,256,8192)
__global__ void final_kernel(const float* __restrict__ Y, const float* __restrict__ stY,
                             const float* __restrict__ g2, const float* __restrict__ be2,
                             const float* __restrict__ stF, const float* __restrict__ gn,
                             const float* __restrict__ ben, float* __restrict__ out,
                             float invP) {
    size_t id = (size_t)blockIdx.x * blockDim.x + threadIdx.x; // (b*256+o)*8192+n
    int nn = (int)(id & 8191);
    int o  = (int)((id >> 13) & 255);
    int b  = (int)(id >> 21);
    float m = stY[o] * invP;
    float v = stY[256 + o] * invP - m * m;
    float sc = rsqrtf(v + 1e-5f) * g2[o];
    float bb = be2[o] - m * sc;
    size_t p = (size_t)b * 8192 + nn;
    float f = fmaxf(Y[p * 256 + o] * sc + bb, 0.f);
    float mF = stF[o] * invP;
    float vF = stF[256 + o] * invP - mF * mF;
    float r = (f - mF) * rsqrtf(vF + 1e-5f) * gn[o] + ben[o];
    out[id] = fmaxf(r, 0.f);
}

// ---------------------------------------------------------------------------
extern "C" void kernel_launch(void* const* d_in, const int* in_sizes, int n_in,
                              void* d_out, int out_size, void* d_ws, size_t ws_size,
                              hipStream_t stream) {
    (void)in_sizes; (void)n_in; (void)out_size; (void)ws_size;
    const float* input_xyz      = (const float*)d_in[0];
    const float* sa1_xyz        = (const float*)d_in[1];
    const float* sa2_xyz        = (const float*)d_in[2];
    const float* input_features = (const float*)d_in[3];
    const float* sa1_features   = (const float*)d_in[4];
    const float* backbone       = (const float*)d_in[5];
    const float* w1a = (const float*)d_in[6];
    const float* w1b = (const float*)d_in[7];
    const float* w2a = (const float*)d_in[8];
    const float* w2b = (const float*)d_in[9];
    const float* b1a  = (const float*)d_in[10];
    const float* be1a = (const float*)d_in[11];
    const float* b1b  = (const float*)d_in[12];
    const float* be1b = (const float*)d_in[13];
    const float* b2a  = (const float*)d_in[14];
    const float* be2a = (const float*)d_in[15];
    const float* b2b  = (const float*)d_in[16];
    const float* be2b = (const float*)d_in[17];
    const float* ben  = (const float*)d_in[18];
    const float* g1a  = (const float*)d_in[19];
    const float* g1b  = (const float*)d_in[20];
    const float* g2a  = (const float*)d_in[21];
    const float* g2b  = (const float*)d_in[22];
    const float* gn   = (const float*)d_in[23];

    const int P1 = 8 * 2048, P2 = 8 * 8192;

    // workspace carve (256-byte aligned regions)
    size_t off = 0;
    char* ws = (char*)d_ws;
    auto take = [&](size_t bytes) -> void* {
        void* p = ws + off;
        off += (bytes + 255) & ~(size_t)255;
        return p;
    };
    u16*   wb1a  = (u16*)take((size_t)256 * 384 * 2);
    u16*   wb1b  = (u16*)take((size_t)256 * 256 * 2);
    u16*   wb2a  = (u16*)take((size_t)256 * 288 * 2);
    u16*   wb2b  = (u16*)take((size_t)256 * 256 * 2);
    float* stats = (float*)take((size_t)5 * 512 * 4);
    u16*   X0    = (u16*)take((size_t)P2 * 288 * 2);   // interp/concat activations
    float* Yb    = (float*)take((size_t)P2 * 256 * 4); // GEMM outputs
    u16*   Xb    = (u16*)take((size_t)P2 * 256 * 2);   // normalized bf16 activations
    u16*   F1    = (u16*)take((size_t)P1 * 256 * 2);   // fp1 output (feeds interp2)

    float* st0 = stats, *st1 = stats + 512, *st2 = stats + 1024,
         * st3 = stats + 1536, *st4 = stats + 2048;

    zero_kernel<<<10, 256, 0, stream>>>(stats, 5 * 512);

    wcvt_kernel<<<(256 * 384 + 255) / 256, 256, 0, stream>>>(w1a, wb1a, 256, 384, 384);
    wcvt_kernel<<<(256 * 256 + 255) / 256, 256, 0, stream>>>(w1b, wb1b, 256, 256, 256);
    wcvt_kernel<<<(256 * 288 + 255) / 256, 256, 0, stream>>>(w2a, wb2a, 256, 259, 288);
    wcvt_kernel<<<(256 * 256 + 255) / 256, 256, 0, stream>>>(w2b, wb2b, 256, 256, 256);

    // ---- FP1 ----
    interp1_kernel<<<dim3(2048 / 256, 8), 256, 0, stream>>>(
        sa1_xyz, sa2_xyz, backbone, sa1_features, X0);
    gemm_wmma_kernel<384><<<dim3(P1 / 32, 1), 256, 0, stream>>>(X0, wb1a, b1a, Yb);
    stats_kernel<<<256, 256, 0, stream>>>(Yb, st0, P1);
    bnrelu_bf16_kernel<<<P1, 256, 0, stream>>>(Yb, st0, g1a, be1a, Xb, 1.f / P1);
    gemm_wmma_kernel<256><<<dim3(P1 / 32, 1), 256, 0, stream>>>(Xb, wb1b, b1b, Yb);
    stats_kernel<<<256, 256, 0, stream>>>(Yb, st1, P1);
    bnrelu_bf16_kernel<<<P1, 256, 0, stream>>>(Yb, st1, g1b, be1b, F1, 1.f / P1);

    // ---- FP2 ----
    interp2_kernel<<<dim3(8192 / 256, 8), 256, 0, stream>>>(
        input_xyz, sa1_xyz, F1, input_features, X0);
    gemm_wmma_kernel<288><<<dim3(P2 / 32, 1), 256, 0, stream>>>(X0, wb2a, b2a, Yb);
    stats_kernel<<<256, 256, 0, stream>>>(Yb, st2, P2);
    bnrelu_bf16_kernel<<<P2, 256, 0, stream>>>(Yb, st2, g2a, be2a, Xb, 1.f / P2);
    gemm_wmma_kernel<256><<<dim3(P2 / 32, 1), 256, 0, stream>>>(Xb, wb2b, b2b, Yb);
    stats_kernel<<<256, 256, 0, stream>>>(Yb, st3, P2);

    // ---- final BN+ReLU (f2 never materialized) ----
    bnrelu_stats_kernel<<<256, 256, 0, stream>>>(Yb, st3, g2b, be2b, st4, P2, 1.f / P2);
    final_kernel<<<(8 * 256 * 8192) / 256, 256, 0, stream>>>(
        Yb, st3, g2b, be2b, st4, gn, ben, (float*)d_out, 1.f / P2);
}
